// MambaSeq2OneBlocks_36369783063163
// MI455X (gfx1250) — compile-verified, matrix operand order checked by reading, not run on hardware
//
#include <hip/hip_runtime.h>
#include <hip/hip_bf16.h>

// ---------------------------------------------------------------------------
// MambaSeq2OneBlocks for MI455X (gfx1250, wave32, WMMA)
//
// Shapes: B=8, C_IN=80, L=4096, D_MODEL=256, D_INNER=512, D_STATE=2,
//         D_CONV=7, DT_RANK=16, N_BLOCK=4.
//
// All GEMMs -> v_wmma_f32_16x16x32_bf16 (RNE f32->bf16 inputs, f32 accum),
// 128x64 block tile, 8 waves (4x2), 32x32 per wave = 4 WMMA / K-step,
// double-buffered LDS with float4 global staging and b64/b128 LDS traffic.
// Scan: 1 thread per (b, d) channel, n=2 states in registers, fused with
// D-skip + silu(res) gating; output written over the dead xi_raw half of xz.
// ---------------------------------------------------------------------------

#define MB_B       8
#define MB_CIN     80
#define MB_L       4096
#define MB_DMODEL  256
#define MB_DINNER  512
#define MB_DSTATE  2
#define MB_DCONV   7
#define MB_DTRANK  16
#define MB_NBLOCK  4
#define MB_ROWS    (MB_B * MB_L)              // 32768 token rows
#define MB_XDBL    (MB_DTRANK + 2*MB_DSTATE)  // 20

typedef __attribute__((ext_vector_type(16))) __bf16 v16bf;
typedef __attribute__((ext_vector_type(8)))  float  v8f;

union Frag16 { v16bf v; unsigned int w[8]; };

// round-to-nearest-even f32 -> bf16, packed pair
__device__ __forceinline__ unsigned int pack_bf16x2(float lo, float hi) {
  unsigned int a = __float_as_uint(lo);
  unsigned int b = __float_as_uint(hi);
  a = (a + 0x7FFFu + ((a >> 16) & 1u)) >> 16;
  b = (b + 0x7FFFu + ((b >> 16) & 1u)) >> 16;
  return a | (b << 16);
}

__device__ __forceinline__ float apply_act(float x, int act) {
  if (act == 1) {                      // softplus
    return (x > 20.0f) ? x : log1pf(__expf(x));
  }
  return x;
}

// guarded float4 load: full-vector fast path, element guards only on K edge
__device__ __forceinline__ float4 load_f4_guard(const float* __restrict__ P,
                                                int row, int ld, int k,
                                                int K, int rmax) {
  float4 v = make_float4(0.f, 0.f, 0.f, 0.f);
  if (row < rmax) {
    const float* p = P + (size_t)row * ld + k;
    if (k + 4 <= K) {
      v = *reinterpret_cast<const float4*>(p);
    } else {
      if (k + 0 < K) v.x = p[0];
      if (k + 1 < K) v.y = p[1];
      if (k + 2 < K) v.z = p[2];
      if (k + 3 < K) v.w = p[3];
    }
  }
  return v;
}

// ---------------------------------------------------------------------------
// WMMA GEMM:  C[M,N] = act( A[M,K] @ W[N,K]^T + bias[N] )
// A row-major, leading dim lda; W compact N x K; C row-major, leading dim ldc.
// Block: 256 threads (8 waves), tile 128x64, K-step 32, double-buffered LDS.
// ---------------------------------------------------------------------------
#define GBM 128
#define GBN 64
#define GBK 32
#define GBK2 (GBK / 2)      // u32 per LDS row

// fragment load: A-frag row r / B-frag col r; element-pair j ->
//   k-pair index = j + 4*g (+4 for j>=4); two b128 loads per fragment
__device__ __forceinline__ void ld_frag(const unsigned int* buf, int r, int g,
                                        Frag16& f) {
  const unsigned int* p = buf + r * GBK2 + 4 * g;
  #pragma unroll
  for (int j = 0; j < 4; ++j) f.w[j] = p[j];
  #pragma unroll
  for (int j = 0; j < 4; ++j) f.w[4 + j] = p[8 + j];
}

__global__ __launch_bounds__(256)
void gemm_wmma_bf16(const float* __restrict__ A, int lda,
                    const float* __restrict__ W,
                    const float* __restrict__ bias,
                    float* __restrict__ C, int ldc,
                    int M, int N, int K, int act)
{
  __shared__ unsigned int As[2][GBM * GBK2];   // 2 x 8KB (bf16 pairs)
  __shared__ unsigned int Bs[2][GBN * GBK2];   // 2 x 4KB

  const int tid  = threadIdx.x;
  const int lane = tid & 31;
  const int wid  = tid >> 5;              // 0..7
  const int g    = (lane >> 4) & 1;       // lane half-group
  const int mr   = lane & 15;

  const int bm = blockIdx.x * GBM;
  const int bn = blockIdx.y * GBN;
  const int wrow = (wid & 3) * 32;        // wave row offset in tile
  const int wcol = (wid >> 2) * 32;       // wave col offset in tile

  v8f acc[2][2] = {};
  float4 ra[4];                           // A: 128*32/4/256 = 4 float4/thread
  float4 rb[2];                           // B: 64*32/4/256  = 2 float4/thread

  // ---- prologue: stage k0 = 0 tile into buffer 0 ----
  #pragma unroll
  for (int c = 0; c < 4; ++c) {
    const int f = tid + c * 256;          // float4 index over A tile
    ra[c] = load_f4_guard(A, bm + (f >> 3), lda, (f & 7) * 4, K, M);
  }
  #pragma unroll
  for (int c = 0; c < 2; ++c) {
    const int f = tid + c * 256;
    rb[c] = load_f4_guard(W, bn + (f >> 3), K, (f & 7) * 4, K, N);
  }
  #pragma unroll
  for (int c = 0; c < 4; ++c) {
    const int f = tid + c * 256;
    uint2 p; p.x = pack_bf16x2(ra[c].x, ra[c].y);
             p.y = pack_bf16x2(ra[c].z, ra[c].w);
    *reinterpret_cast<uint2*>(&As[0][(f >> 3) * GBK2 + (f & 7) * 2]) = p;
  }
  #pragma unroll
  for (int c = 0; c < 2; ++c) {
    const int f = tid + c * 256;
    uint2 p; p.x = pack_bf16x2(rb[c].x, rb[c].y);
             p.y = pack_bf16x2(rb[c].z, rb[c].w);
    *reinterpret_cast<uint2*>(&Bs[0][(f >> 3) * GBK2 + (f & 7) * 2]) = p;
  }
  __syncthreads();

  int cur = 0;
  for (int k0 = 0; k0 < K; k0 += GBK) {
    const bool has_next = (k0 + GBK) < K;

    // ---- issue global loads for the NEXT tile (latency hidden by WMMA) ----
    if (has_next) {
      const int kn = k0 + GBK;
      #pragma unroll
      for (int c = 0; c < 4; ++c) {
        const int f = tid + c * 256;
        ra[c] = load_f4_guard(A, bm + (f >> 3), lda, kn + (f & 7) * 4, K, M);
      }
      #pragma unroll
      for (int c = 0; c < 2; ++c) {
        const int f = tid + c * 256;
        rb[c] = load_f4_guard(W, bn + (f >> 3), K, kn + (f & 7) * 4, K, N);
      }
      // prefetch tile k0 + 2*GBK of the activation stream
      if (k0 + 2 * GBK < K) {
        const int pr = bm + (tid >> 1);
        if (pr < M)
          __builtin_prefetch((const void*)(A + (size_t)pr * lda + k0 + 2 * GBK), 0, 3);
      }
    }

    // ---- compute on current buffer: 4 x v_wmma_f32_16x16x32_bf16 ----
    Frag16 fa0, fa1, fb0, fb1;
    ld_frag(As[cur], wrow + mr,      g, fa0);
    ld_frag(As[cur], wrow + 16 + mr, g, fa1);
    ld_frag(Bs[cur], wcol + mr,      g, fb0);
    ld_frag(Bs[cur], wcol + 16 + mr, g, fb1);

    acc[0][0] = __builtin_amdgcn_wmma_f32_16x16x32_bf16(
        false, fa0.v, false, fb0.v, (short)0, acc[0][0], false, false);
    acc[0][1] = __builtin_amdgcn_wmma_f32_16x16x32_bf16(
        false, fa0.v, false, fb1.v, (short)0, acc[0][1], false, false);
    acc[1][0] = __builtin_amdgcn_wmma_f32_16x16x32_bf16(
        false, fa1.v, false, fb0.v, (short)0, acc[1][0], false, false);
    acc[1][1] = __builtin_amdgcn_wmma_f32_16x16x32_bf16(
        false, fa1.v, false, fb1.v, (short)0, acc[1][1], false, false);

    // ---- store staged registers into the alternate buffer ----
    if (has_next) {
      const int nxt = cur ^ 1;
      #pragma unroll
      for (int c = 0; c < 4; ++c) {
        const int f = tid + c * 256;
        uint2 p; p.x = pack_bf16x2(ra[c].x, ra[c].y);
                 p.y = pack_bf16x2(ra[c].z, ra[c].w);
        *reinterpret_cast<uint2*>(&As[nxt][(f >> 3) * GBK2 + (f & 7) * 2]) = p;
      }
      #pragma unroll
      for (int c = 0; c < 2; ++c) {
        const int f = tid + c * 256;
        uint2 p; p.x = pack_bf16x2(rb[c].x, rb[c].y);
                 p.y = pack_bf16x2(rb[c].z, rb[c].w);
        *reinterpret_cast<uint2*>(&Bs[nxt][(f >> 3) * GBK2 + (f & 7) * 2]) = p;
      }
    }
    __syncthreads();   // single barrier per iteration (buffers alternate)
    cur ^= 1;
  }

  // ---- epilogue: D layout lane->col (mr), VGPR v -> row (v + 8*g) ----
  #pragma unroll
  for (int ti = 0; ti < 2; ++ti) {
    #pragma unroll
    for (int v = 0; v < 8; ++v) {
      const int m = bm + wrow + ti * 16 + v + 8 * g;
      if (m >= M) continue;
      #pragma unroll
      for (int tj = 0; tj < 2; ++tj) {
        const int n = bn + wcol + tj * 16 + mr;
        if (n < N) {
          float r = acc[ti][tj][v] + (bias ? bias[n] : 0.0f);
          C[(size_t)m * ldc + n] = apply_act(r, act);
        }
      }
    }
  }
}

// ---------------------------------------------------------------------------
// Pack x (B, C_IN, L) -> xT (B*L, C_IN) for the input projection GEMM.
// ---------------------------------------------------------------------------
__global__ void pack_xT_kernel(const float* __restrict__ x,
                               float* __restrict__ xT)
{
  size_t idx = (size_t)blockIdx.x * blockDim.x + threadIdx.x;
  const size_t total = (size_t)MB_ROWS * MB_CIN;
  if (idx >= total) return;
  const int c  = (int)(idx % MB_CIN);
  const size_t bl = idx / MB_CIN;         // b*L + l
  const int l  = (int)(bl & (MB_L - 1));
  const int b  = (int)(bl >> 12);         // L = 4096 = 2^12
  xT[idx] = x[((size_t)b * MB_CIN + c) * MB_L + l];
}

// ---------------------------------------------------------------------------
// Causal depthwise conv (d_conv=7) + bias + silu, 4 channels per thread.
// xz layout: (B*L, 1024); xi_raw = cols [0,512), res = cols [512,1024).
// ---------------------------------------------------------------------------
__global__ void conv_silu_kernel(const float* __restrict__ xz,
                                 const float* __restrict__ cw,   // (512,7)
                                 const float* __restrict__ cb,   // (512)
                                 float* __restrict__ xi)
{
  size_t idx = (size_t)blockIdx.x * blockDim.x + threadIdx.x;
  const size_t total = (size_t)MB_ROWS * (MB_DINNER / 4);
  if (idx >= total) return;
  const int d4 = (int)(idx & (MB_DINNER / 4 - 1)) * 4;   // base channel
  const size_t bl = idx >> 7;                            // b*L + l
  const int l = (int)(bl & (MB_L - 1));

  float4 acc = make_float4(cb[d4 + 0], cb[d4 + 1], cb[d4 + 2], cb[d4 + 3]);
  #pragma unroll
  for (int k = 0; k < MB_DCONV; ++k) {
    const int ll = l - (MB_DCONV - 1) + k;
    if (ll >= 0) {
      const float4 v = *reinterpret_cast<const float4*>(
          xz + (bl - (MB_DCONV - 1) + k) * (2 * MB_DINNER) + d4);
      acc.x += cw[(d4 + 0) * MB_DCONV + k] * v.x;
      acc.y += cw[(d4 + 1) * MB_DCONV + k] * v.y;
      acc.z += cw[(d4 + 2) * MB_DCONV + k] * v.z;
      acc.w += cw[(d4 + 3) * MB_DCONV + k] * v.w;
    }
  }
  acc.x = acc.x / (1.0f + __expf(-acc.x));
  acc.y = acc.y / (1.0f + __expf(-acc.y));
  acc.z = acc.z / (1.0f + __expf(-acc.z));
  acc.w = acc.w / (1.0f + __expf(-acc.w));
  *reinterpret_cast<float4*>(xi + bl * MB_DINNER + d4) = acc;
}

// ---------------------------------------------------------------------------
// Selective scan fused with D-skip and silu(res) gating.
// One thread per (b, d) channel; n = 2 states in registers.
// Writes y into xz cols [0,512) (xi_raw half is dead after the conv).
// ---------------------------------------------------------------------------
__global__ void scan_kernel(const float* __restrict__ delta,  // (B*L, 512)
                            const float* __restrict__ xdbl,   // (B*L, 20)
                            const float* __restrict__ xi,     // (B*L, 512)
                            float* __restrict__ xz,           // (B*L, 1024)
                            const float* __restrict__ A_log,  // (512, 2)
                            const float* __restrict__ Dp)     // (512)
{
  const int idx = blockIdx.x * blockDim.x + threadIdx.x;
  if (idx >= MB_B * MB_DINNER) return;
  const int b = idx >> 9;
  const int d = idx & (MB_DINNER - 1);

  const float A0 = -__expf(A_log[d * 2 + 0]);
  const float A1 = -__expf(A_log[d * 2 + 1]);
  const float Dd = Dp[d];

  float h0 = 0.0f, h1 = 0.0f;
  const size_t base = (size_t)b * MB_L;
  for (int l = 0; l < MB_L; ++l) {
    const size_t row = base + l;
    const float dt = delta[row * MB_DINNER + d];
    const float u  = xi[row * MB_DINNER + d];
    const float* xd = xdbl + row * MB_XDBL;
    const float Bm0 = xd[MB_DTRANK + 0];
    const float Bm1 = xd[MB_DTRANK + 1];
    const float Cm0 = xd[MB_DTRANK + 2];
    const float Cm1 = xd[MB_DTRANK + 3];

    h0 = __expf(dt * A0) * h0 + dt * Bm0 * u;
    h1 = __expf(dt * A1) * h1 + dt * Bm1 * u;

    float y = h0 * Cm0 + h1 * Cm1 + u * Dd;
    const float r = xz[row * (2 * MB_DINNER) + MB_DINNER + d];
    y *= r / (1.0f + __expf(-r));          // y * silu(res)
    xz[row * (2 * MB_DINNER) + d] = y;
  }
}

// ---------------------------------------------------------------------------
// Final mean over sequence: out[b, c] = mean_l h[b, l, c]
// ---------------------------------------------------------------------------
__global__ void mean_kernel(const float* __restrict__ h,
                            float* __restrict__ out)
{
  const int b = blockIdx.x;        // 8
  const int c = threadIdx.x;       // 256
  float s = 0.0f;
  for (int l = 0; l < MB_L; ++l)
    s += h[((size_t)b * MB_L + l) * MB_DMODEL + c];
  out[b * MB_DMODEL + c] = s * (1.0f / MB_L);
}

// ---------------------------------------------------------------------------
// Host-side orchestration
// ---------------------------------------------------------------------------
extern "C" void kernel_launch(void* const* d_in, const int* in_sizes, int n_in,
                              void* d_out, int out_size, void* d_ws, size_t ws_size,
                              hipStream_t stream) {
  (void)in_sizes; (void)n_in; (void)out_size; (void)ws_size;

  const float* x       = (const float*)d_in[0];   // (8, 80, 4096)
  const float* proj_w  = (const float*)d_in[1];   // (256, 80)
  const float* proj_b  = (const float*)d_in[2];   // (256)
  const float* in_w    = (const float*)d_in[3];   // (4, 1024, 256)
  const float* conv_w  = (const float*)d_in[4];   // (4, 512, 7)
  const float* conv_b  = (const float*)d_in[5];   // (4, 512)
  const float* xproj_w = (const float*)d_in[6];   // (4, 20, 512)
  const float* dt_w    = (const float*)d_in[7];   // (4, 512, 16)
  const float* dt_b    = (const float*)d_in[8];   // (4, 512)
  const float* A_log   = (const float*)d_in[9];   // (4, 512, 2)
  const float* Dparam  = (const float*)d_in[10];  // (4, 512)
  const float* out_w   = (const float*)d_in[11];  // (4, 256, 512)
  float* out = (float*)d_out;                     // (8, 256)

  // Workspace carve-up (floats)
  float* ws   = (float*)d_ws;
  float* xT    = ws;  ws += (size_t)MB_ROWS * MB_CIN;        //  10.5 MB
  float* hA    = ws;  ws += (size_t)MB_ROWS * MB_DMODEL;     //  33.6 MB
  float* hB    = ws;  ws += (size_t)MB_ROWS * MB_DMODEL;     //  33.6 MB
  float* xz    = ws;  ws += (size_t)MB_ROWS * 2 * MB_DINNER; // 134.2 MB
  float* xi    = ws;  ws += (size_t)MB_ROWS * MB_DINNER;     //  67.1 MB
  float* xdbl  = ws;  ws += (size_t)MB_ROWS * MB_XDBL;       //   2.6 MB
  float* delta = ws;  ws += (size_t)MB_ROWS * MB_DINNER;     //  67.1 MB

  // 1) pack x -> xT, then input projection GEMM (M=32768, N=256, K=80)
  {
    const size_t total = (size_t)MB_ROWS * MB_CIN;
    pack_xT_kernel<<<(unsigned)((total + 255) / 256), 256, 0, stream>>>(x, xT);
    dim3 g((MB_ROWS + GBM - 1) / GBM, (MB_DMODEL + GBN - 1) / GBN);
    gemm_wmma_bf16<<<g, 256, 0, stream>>>(xT, MB_CIN, proj_w, proj_b,
                                          hA, MB_DMODEL,
                                          MB_ROWS, MB_DMODEL, MB_CIN, 0);
  }

  float* hcur = hA;
  float* hnxt = hB;

  for (int i = 0; i < MB_NBLOCK; ++i) {
    const float* in_w_i    = in_w    + (size_t)i * 2 * MB_DINNER * MB_DMODEL;
    const float* conv_w_i  = conv_w  + (size_t)i * MB_DINNER * MB_DCONV;
    const float* conv_b_i  = conv_b  + (size_t)i * MB_DINNER;
    const float* xproj_w_i = xproj_w + (size_t)i * MB_XDBL * MB_DINNER;
    const float* dt_w_i    = dt_w    + (size_t)i * MB_DINNER * MB_DTRANK;
    const float* dt_b_i    = dt_b    + (size_t)i * MB_DINNER;
    const float* A_log_i   = A_log   + (size_t)i * MB_DINNER * MB_DSTATE;
    const float* D_i       = Dparam  + (size_t)i * MB_DINNER;
    const float* out_w_i   = out_w   + (size_t)i * MB_DMODEL * MB_DINNER;

    // in-projection: xz = h @ in_w^T   (M=32768, N=1024, K=256)
    {
      dim3 g((MB_ROWS + GBM - 1) / GBM, (2 * MB_DINNER + GBN - 1) / GBN);
      gemm_wmma_bf16<<<g, 256, 0, stream>>>(hcur, MB_DMODEL, in_w_i, nullptr,
                                            xz, 2 * MB_DINNER,
                                            MB_ROWS, 2 * MB_DINNER, MB_DMODEL, 0);
    }
    // causal depthwise conv + silu (float4 over channels)
    {
      const size_t total = (size_t)MB_ROWS * (MB_DINNER / 4);
      conv_silu_kernel<<<(unsigned)((total + 255) / 256), 256, 0, stream>>>(
          xz, conv_w_i, conv_b_i, xi);
    }
    // x-projection: x_dbl = xi @ xproj_w^T  (M=32768, N=20, K=512)
    {
      dim3 g((MB_ROWS + GBM - 1) / GBM, (MB_XDBL + GBN - 1) / GBN);
      gemm_wmma_bf16<<<g, 256, 0, stream>>>(xi, MB_DINNER, xproj_w_i, nullptr,
                                            xdbl, MB_XDBL,
                                            MB_ROWS, MB_XDBL, MB_DINNER, 0);
    }
    // dt: delta = softplus(x_dbl[:, :16] @ dt_w^T + dt_b)  (N=512, K=16)
    {
      dim3 g((MB_ROWS + GBM - 1) / GBM, (MB_DINNER + GBN - 1) / GBN);
      gemm_wmma_bf16<<<g, 256, 0, stream>>>(xdbl, MB_XDBL, dt_w_i, dt_b_i,
                                            delta, MB_DINNER,
                                            MB_ROWS, MB_DINNER, MB_DTRANK, 1);
    }
    // fused selective scan (+ D skip, silu(res) gate); y -> xz[:, 0:512)
    {
      scan_kernel<<<(MB_B * MB_DINNER + 255) / 256, 256, 0, stream>>>(
          delta, xdbl, xi, xz, A_log_i, D_i);
    }
    // out-projection: h_next = y @ out_w^T  (M=32768, N=256, K=512)
    {
      dim3 g((MB_ROWS + GBM - 1) / GBM, (MB_DMODEL + GBN - 1) / GBN);
      gemm_wmma_bf16<<<g, 256, 0, stream>>>(xz, 2 * MB_DINNER, out_w_i, nullptr,
                                            hnxt, MB_DMODEL,
                                            MB_ROWS, MB_DMODEL, MB_DINNER, 0);
    }
    float* t = hcur; hcur = hnxt; hnxt = t;
  }

  // final mean over sequence
  mean_kernel<<<MB_B, MB_DMODEL, 0, stream>>>(hcur, out);
}
